// Random_Attention_70111046140183
// MI455X (gfx1250) — matrix operand order repaired and checked by reading, not compiled, not run on hardware
//
#include <hip/hip_runtime.h>

typedef __attribute__((ext_vector_type(16))) __bf16 v16bf;
typedef __attribute__((ext_vector_type(8)))  float  v8f;
typedef unsigned short u16;
typedef unsigned int   u32;

#define S_LEN 2048
#define D_DIM 1024
#define NH    16
#define DH    64
#define BATCH 4
#define MROWS (BATCH * S_LEN)
#define GSTRIDE 72    // GEMM LDS row stride in u16 (64 data + 8 pad), 16B aligned
#define PSTRIDE 40    // attention P-tile LDS row stride in u16 (32 data + 8 pad)

// ---------- helpers ----------

__device__ __forceinline__ u16 f2bf(float f) {
  u32 u = __float_as_uint(f);
  u32 r = u + 0x7FFFu + ((u >> 16) & 1u);   // round-to-nearest-even
  return (u16)(r >> 16);
}

union Frag {
  v16bf v;
  u32   u[8];
};

// 16-bit A-matrix 16x32 fragment (ISA 7.12.2): lane m = lane&15, hf = lane>>4.
// VGPR0-3 hold K = 8*hf..8*hf+7 (packed pairs), VGPR4-7 hold K = 16+8*hf..23+8*hf.
__device__ __forceinline__ v16bf load_a_frag(const u16* row, int hf) {
  Frag f;
  const u32* p0 = (const u32*)(row + 8 * hf);
  const u32* p1 = (const u32*)(row + 16 + 8 * hf);
#pragma unroll
  for (int i = 0; i < 4; ++i) { f.u[i] = p0[i]; f.u[4 + i] = p1[i]; }
  return f.v;
}

// 16-bit B-matrix 32x16 fragment: lane n = lane&15; VGPR0-3 hold K = 8*hf..8*hf+7,
// VGPR4-7 hold K = 16+8*hf..23+8*hf. `row` points at the K-contiguous B^T row.
__device__ __forceinline__ v16bf load_b_frag(const u16* row, int hf) {
  Frag f;
  const u32* p0 = (const u32*)(row + 8 * hf);
  const u32* p1 = (const u32*)(row + 16 + 8 * hf);
#pragma unroll
  for (int i = 0; i < 4; ++i) { f.u[i] = p0[i]; f.u[4 + i] = p1[i]; }
  return f.v;
}

__device__ __forceinline__ v8f wmma_bf16(v16bf a, v16bf b, v8f c) {
  return __builtin_amdgcn_wmma_f32_16x16x32_bf16(false, a, false, b, (short)0, c,
                                                 false, false);
}

// Async global->LDS copy of 16 bytes (per lane), ASYNCcnt-tracked (CDNA5 TDM-lite).
__device__ __forceinline__ void async_b128(u32 lds_byte, const u16* gbase, u32 gbyte) {
  asm volatile("global_load_async_to_lds_b128 %0, %1, %2"
               :: "v"(lds_byte), "v"(gbyte), "s"(gbase)
               : "memory");
}

// Stage a 128x64 bf16 tile: G[(grow0+r)*1024 + kk + c] -> lbase[r*GSTRIDE + c].
// 256 threads, 4 async b128 per thread.
__device__ __forceinline__ void stage64(const u16* __restrict__ G, int grow0, int kk,
                                        u16* lbase, int t) {
  const int r  = t >> 1;
  const int c0 = (t & 1) * 32;
  u32 lds = (u32)(uintptr_t)(lbase + (u32)(r * GSTRIDE + c0));
  u32 gb  = ((u32)(grow0 + r) * (u32)D_DIM + (u32)(kk + c0)) * 2u;
#pragma unroll
  for (int j = 0; j < 4; ++j)
    async_b128(lds + j * 16u, G, gb + j * 16u);
}

// ---------- elementwise f32 -> bf16 ----------

__global__ __launch_bounds__(256)
void cvt_bf16_kernel(const float* __restrict__ src, u16* __restrict__ dst) {
  int i = blockIdx.x * 256 + threadIdx.x;
  dst[i] = f2bf(src[i]);
}

// ---------- weight convert + transpose: W f32 [K][N] -> WT bf16 [N][K] ----------

__global__ __launch_bounds__(256)
void transpose_w_kernel(const float* __restrict__ W, u16* __restrict__ WT) {
  int idx = blockIdx.x * 256 + threadIdx.x;
  int n = idx >> 10;
  int k = idx & 1023;
  WT[idx] = f2bf(W[(size_t)k * D_DIM + n]);
}

// ---------- GEMM: out = A(bf16 [M][K]) * B(via B^T bf16 [N][K]) + bias ----------
// MODE 0: Q -> bf16 [B,H,S,DH]   MODE 1: K -> bf16 [B,H,S,DH]
// MODE 2: V -> bf16 [B,H,DH,S]   MODE 3: f32 [M][N] (output projection)

template <int MODE>
__global__ __launch_bounds__(256)
void gemm_kernel(const u16* __restrict__ A, const u16* __restrict__ BT,
                 const float* __restrict__ bias,
                 u16* __restrict__ out16, float* __restrict__ out32) {
  __shared__ u16 Asub[2][128 * GSTRIDE];
  __shared__ u16 Bsub[2][128 * GSTRIDE];

  const int t    = threadIdx.x;
  const int lane = t & 31;
  const int hf   = (t >> 4) & 1;
  const int wid  = t >> 5;
  const int wm0  = (wid & 3) * 32;   // 4 wave-rows of 32
  const int wn0  = (wid >> 2) * 64;  // 2 wave-cols of 64
  const int row0 = blockIdx.x * 128;
  const int col0 = blockIdx.y * 128;
  const int ln   = lane & 15;

  v8f acc[2][4] = {};

  // prologue stage into buffer 0 (8 async ops per thread)
  stage64(A,  row0, 0, &Asub[0][0], t);
  stage64(BT, col0, 0, &Bsub[0][0], t);

  for (int kk = 0; kk < D_DIM; kk += 64) {
    const int buf = (kk >> 6) & 1;
    if (kk + 64 < D_DIM) {
      stage64(A,  row0, kk + 64, &Asub[buf ^ 1][0], t);
      stage64(BT, col0, kk + 64, &Bsub[buf ^ 1][0], t);
      asm volatile("s_wait_asynccnt 0x8" ::: "memory");  // current buf complete
    } else {
      asm volatile("s_wait_asynccnt 0x0" ::: "memory");
    }
    __syncthreads();

#pragma unroll
    for (int kc = 0; kc < 2; ++kc) {
      v16bf af[2];
#pragma unroll
      for (int tm = 0; tm < 2; ++tm)
        af[tm] = load_a_frag(&Asub[buf][(wm0 + tm * 16 + ln) * GSTRIDE + kc * 32], hf);
      v16bf bfm[4];
#pragma unroll
      for (int tn = 0; tn < 4; ++tn)
        bfm[tn] = load_b_frag(&Bsub[buf][(wn0 + tn * 16 + ln) * GSTRIDE + kc * 32], hf);
#pragma unroll
      for (int tm = 0; tm < 2; ++tm)
#pragma unroll
        for (int tn = 0; tn < 4; ++tn)
          acc[tm][tn] = wmma_bf16(af[tm], bfm[tn], acc[tm][tn]);
    }
    __syncthreads();   // safe to overwrite buf next iteration
  }

  // epilogue (MODE folded at compile time)
#pragma unroll
  for (int tm = 0; tm < 2; ++tm)
#pragma unroll
    for (int tn = 0; tn < 4; ++tn) {
      const int c  = col0 + wn0 + tn * 16 + ln;
      const float bv = bias[c];
#pragma unroll
      for (int v = 0; v < 8; ++v) {
        const int r = row0 + wm0 + tm * 16 + v + 8 * hf;  // token
        const float val = acc[tm][tn][v] + bv;
        if (MODE == 3) {
          out32[(size_t)r * D_DIM + c] = val;
        } else {
          const int b = r >> 11, s = r & (S_LEN - 1);
          const int h = c >> 6,  dh = c & (DH - 1);
          size_t idx;
          if (MODE == 2)
            idx = (((size_t)(b * NH + h) * DH + dh) << 11) + s;   // [B,H,DH,S]
          else
            idx = (((size_t)(b * NH + h) << 11) + s) * DH + dh;   // [B,H,S,DH]
          out16[idx] = f2bf(val);
        }
      }
    }
}

// ---------- flash attention: one wave per 16-query tile ----------

__global__ __launch_bounds__(256)
void attn128_kernel(const u16* __restrict__ Qg, const u16* __restrict__ Kg,
                    const u16* __restrict__ Vt, const unsigned char* __restrict__ mask,
                    u16* __restrict__ ctxb) {
  __shared__ u16 Plds[8][16 * PSTRIDE];   // wave-private P re-layout tiles

  const int t    = threadIdx.x;
  const int lane = t & 31;
  const int hf   = (t >> 4) & 1;
  const int wid  = t >> 5;
  const int bh   = blockIdx.y;             // b*NH + h
  const int b    = bh >> 4;
  const int h    = bh & 15;
  const int q0   = blockIdx.x * 128 + wid * 16;
  const int ln   = lane & 15;
  const float scale = 0.125f;              // 1/sqrt(DH)

  // Q fragments (16 queries x 64 dh) resident for the whole sweep
  const u16* Qrow = Qg + ((size_t)bh * S_LEN + q0 + ln) * DH;
  v16bf qf[2];
#pragma unroll
  for (int c = 0; c < 2; ++c) qf[c] = load_a_frag(Qrow + c * 32, hf);

  v8f o[4] = {};
  float mr[8], lr[8];
#pragma unroll
  for (int v = 0; v < 8; ++v) { mr[v] = -3.0e38f; lr[v] = 0.0f; }

  u16* Pw = &Plds[wid][0];

  for (int kc = 0; kc < S_LEN; kc += 32) {
    // prefetch next K/V chunk rows (L2 streaming hint -> global_prefetch_b8)
    if (kc + 32 < S_LEN) {
      __builtin_prefetch(Kg + ((size_t)bh * S_LEN + kc + 32 + ln) * DH, 0, 1);
      __builtin_prefetch(Vt + ((size_t)bh * DH + ln) * S_LEN + kc + 32, 0, 1);
    }

    // scores S = Q K^T for 16 q x 32 keys
    v8f sc2[2] = {};
#pragma unroll
    for (int tn = 0; tn < 2; ++tn) {
      const u16* Krow = Kg + ((size_t)bh * S_LEN + kc + tn * 16 + ln) * DH;
#pragma unroll
      for (int c = 0; c < 2; ++c) {
        v16bf kb = load_b_frag(Krow + c * 32, hf);
        sc2[tn] = wmma_bf16(qf[c], kb, sc2[tn]);
      }
    }

    // mask + scale
    float sval[2][8];
#pragma unroll
    for (int tn = 0; tn < 2; ++tn)
#pragma unroll
      for (int v = 0; v < 8; ++v) {
        int q   = q0 + v + 8 * hf;
        int key = kc + tn * 16 + ln;
        float sv = sc2[tn][v] * scale;
        sval[tn][v] = mask[(size_t)q * S_LEN + key] ? sv : -1.0e30f;
      }

    // row max across the 16 lanes holding this row's keys
    float mc[8];
#pragma unroll
    for (int v = 0; v < 8; ++v) mc[v] = fmaxf(sval[0][v], sval[1][v]);
#pragma unroll
    for (int off = 8; off >= 1; off >>= 1)
#pragma unroll
      for (int v = 0; v < 8; ++v)
        mc[v] = fmaxf(mc[v], __shfl_xor(mc[v], off, 32));

    // online softmax update; stash P (bf16) into wave-private LDS tile
    float al[8];
#pragma unroll
    for (int v = 0; v < 8; ++v) {
      float mn = fmaxf(mr[v], mc[v]);
      al[v] = __expf(mr[v] - mn);
      mr[v] = mn;
      float p0 = __expf(sval[0][v] - mn);
      float p1 = __expf(sval[1][v] - mn);
      Pw[(v + 8 * hf) * PSTRIDE + ln]      = f2bf(p0);
      Pw[(v + 8 * hf) * PSTRIDE + 16 + ln] = f2bf(p1);
      float ps = p0 + p1;
#pragma unroll
      for (int off = 8; off >= 1; off >>= 1) ps += __shfl_xor(ps, off, 32);
      lr[v] = lr[v] * al[v] + ps;
    }

    // rescale O accumulators
#pragma unroll
    for (int tile = 0; tile < 4; ++tile)
#pragma unroll
      for (int v = 0; v < 8; ++v) o[tile][v] *= al[v];

    // O += P V  (P re-loaded in A layout; V pre-transposed [B,H,DH,S])
    v16bf pf = load_a_frag(&Pw[ln * PSTRIDE], hf);
#pragma unroll
    for (int tile = 0; tile < 4; ++tile) {
      const u16* Vrow = Vt + ((size_t)bh * DH + tile * 16 + ln) * S_LEN + kc;
      v16bf vb = load_b_frag(Vrow, hf);
      o[tile] = wmma_bf16(pf, vb, o[tile]);
    }
  }

  // normalize and write ctx bf16 [B,S,D] (A-operand of the output projection)
  float inv[8];
#pragma unroll
  for (int v = 0; v < 8; ++v) inv[v] = 1.0f / lr[v];
#pragma unroll
  for (int tile = 0; tile < 4; ++tile)
#pragma unroll
    for (int v = 0; v < 8; ++v) {
      int q = q0 + v + 8 * hf;
      int c = h * DH + tile * 16 + ln;
      ctxb[((size_t)b * S_LEN + q) * D_DIM + c] = f2bf(o[tile][v] * inv[v]);
    }
}

// ---------- launch ----------

extern "C" void kernel_launch(void* const* d_in, const int* in_sizes, int n_in,
                              void* d_out, int out_size, void* d_ws, size_t ws_size,
                              hipStream_t stream) {
  (void)in_sizes; (void)n_in; (void)out_size; (void)ws_size;

  const float*         X    = (const float*)d_in[0];
  const unsigned char* mask = (const unsigned char*)d_in[1];
  const float* Wq = (const float*)d_in[2];
  const float* bq = (const float*)d_in[3];
  const float* Wk = (const float*)d_in[4];
  const float* bk = (const float*)d_in[5];
  const float* Wv = (const float*)d_in[6];
  const float* bv = (const float*)d_in[7];
  const float* Wo = (const float*)d_in[8];
  const float* bo = (const float*)d_in[9];

  char* ws = (char*)d_ws;
  const size_t WTb  = (size_t)D_DIM * D_DIM * sizeof(u16);   // 2 MB each
  u16* WqT = (u16*)(ws + 0 * WTb);
  u16* WkT = (u16*)(ws + 1 * WTb);
  u16* WvT = (u16*)(ws + 2 * WTb);
  u16* WoT = (u16*)(ws + 3 * WTb);
  const size_t QKVe = (size_t)MROWS * D_DIM;                 // 8.39M elems
  u16* Xb   = (u16*)(ws + 4 * WTb);
  u16* Qb   = Xb + QKVe;
  u16* Kb   = Qb + QKVe;
  u16* Vb   = Kb + QKVe;
  u16* ctxb = Vb + QKVe;

  dim3 blk(256);

  cvt_bf16_kernel<<<dim3((MROWS * D_DIM) / 256), blk, 0, stream>>>(X, Xb);

  dim3 gT((D_DIM * D_DIM) / 256);
  transpose_w_kernel<<<gT, blk, 0, stream>>>(Wq, WqT);
  transpose_w_kernel<<<gT, blk, 0, stream>>>(Wk, WkT);
  transpose_w_kernel<<<gT, blk, 0, stream>>>(Wv, WvT);
  transpose_w_kernel<<<gT, blk, 0, stream>>>(Wo, WoT);

  dim3 gG(MROWS / 128, D_DIM / 128);
  gemm_kernel<0><<<gG, blk, 0, stream>>>(Xb, WqT, bq, Qb, nullptr);
  gemm_kernel<1><<<gG, blk, 0, stream>>>(Xb, WkT, bk, Kb, nullptr);
  gemm_kernel<2><<<gG, blk, 0, stream>>>(Xb, WvT, bv, Vb, nullptr);

  dim3 gA(S_LEN / 128, BATCH * NH);
  attn128_kernel<<<gA, blk, 0, stream>>>(Qb, Kb, Vb, mask, ctxb);

  gemm_kernel<3><<<gG, blk, 0, stream>>>(ctxb, WoT, bo, nullptr, (float*)d_out);
}
